// S6_15762529976505
// MI455X (gfx1250) — matrix operand order repaired and checked by reading, not compiled, not run on hardware
//
#include <hip/hip_runtime.h>
#include <hip/hip_bf16.h>
#include <math.h>

// Problem constants (from reference)
#define BATCH   16
#define SEQ     4096
#define DM      256     // d_model
#define NS      16      // n_state
#define KW      3       // conv width

// Tiling
#define MTILE   64                  // seq rows per workgroup
#define NTILES  18                  // 16 delta n-tiles + Bm + Cm
#define KCHUNKS 24                  // im2col K = 3*256 = 768 -> 24 chunks of 32
#define WAVES   NTILES
#define THREADS (WAVES * 32)        // 576
#define XROWS   (MTILE + 2)         // halo rows
#define XSTRIDE 264                 // bf16 elems per LDS row (528B = 33*16B, bank-conflict free)

typedef __bf16 bf16_t;
typedef bf16_t v16bf __attribute__((ext_vector_type(16)));
typedef float  v8f   __attribute__((ext_vector_type(8)));

union Frag {
    v16bf v;
    uint4 q[2];
};

#if __has_builtin(__builtin_amdgcn_sched_group_barrier)
#define SCHED_PIPELINE()                                            \
    do {                                                            \
        __builtin_amdgcn_sched_group_barrier(0x020, 2, 0); /* VMEM rd: B frag */ \
        __builtin_amdgcn_sched_group_barrier(0x100, 8, 0); /* DS rd: 4 A frags */ \
        __builtin_amdgcn_sched_group_barrier(0x008, 4, 0); /* WMMA x4 */ \
    } while (0)
#else
#define SCHED_PIPELINE() do {} while (0)
#endif

// ---------------------------------------------------------------------------
// Prep: convert fp32 conv weights into the exact per-lane bf16 register image
// of WMMA B fragments (32x16 K-major tiles).
// Flat bf16 index = ((tile*KCHUNKS + kc)*32 + lane)*16 + e, where element e of
// lane holds W[k = 32*kc + 16*(lane>>4) + e][n = 16*tile + (lane&15)].
// ---------------------------------------------------------------------------
__global__ void s6_prep_weights(const float* __restrict__ w1,
                                const float* __restrict__ w2,
                                const float* __restrict__ w3,
                                bf16_t* __restrict__ wb)
{
    const int TOT = NTILES * KCHUNKS * 32 * 16;
    int idx = blockIdx.x * blockDim.x + threadIdx.x;
    if (idx >= TOT) return;
    int e    = idx & 15;
    int lane = (idx >> 4) & 31;
    int kc   = (idx >> 9) % KCHUNKS;
    int t    = (idx >> 9) / KCHUNKS;
    int k    = 32 * kc + 16 * (lane >> 4) + e;   // 0..767 im2col row
    int n    = lane & 15;
    int shift = k >> 8;                          // which tap of the width-3 conv
    int d     = k & 255;                         // input channel
    float w;
    if (t < 16)       w = w1[((size_t)shift * DM + d) * DM + 16 * t + n];
    else if (t == 16) w = w2[((size_t)shift * DM + d) * NS + n];
    else              w = w3[((size_t)shift * DM + d) * NS + n];
    wb[idx] = (bf16_t)w;
}

// ---------------------------------------------------------------------------
// Fused kernel: one WG = 64 seq rows of one batch element, 18 waves.
// Wave t<16: delta channels [16t,16t+16); wave 16: Bm; wave 17: Cm.
// ---------------------------------------------------------------------------
__global__ __launch_bounds__(THREADS)
void s6_fused(const float* __restrict__ x,
              const float* __restrict__ b1,
              const float* __restrict__ b2,
              const float* __restrict__ b3,
              const bf16_t* __restrict__ wb,
              float* __restrict__ out)
{
    __shared__ bf16_t xs[XROWS * XSTRIDE];      // 34,848 B
    __shared__ float  bc[2][MTILE][NS];         //  8,192 B
    __shared__ float  sv[MTILE];                //    256 B

    const int b    = blockIdx.x >> 6;           // batch
    const int gRow = (blockIdx.x & 63) * MTILE; // tile start within sequence
    const int tid  = threadIdx.x;
    const int lane = tid & 31;
    const int wave = tid >> 5;                  // n-tile id
    const int kh   = lane >> 4;                 // lane half (WMMA layout)
    const int ln   = lane & 15;

    const float* xb = x + (size_t)b * SEQ * DM;

    // ---- Stage x halo tile into LDS as bf16 ('same' padding -> zeros) ----
    for (int i = tid; i < XROWS * DM; i += THREADS) {
        int r = i >> 8;                         // / DM
        int c = i & (DM - 1);
        int row = gRow + r - 1;
        float v = (row >= 0 && row < SEQ) ? xb[(size_t)row * DM + c] : 0.0f;
        xs[r * XSTRIDE + c] = (bf16_t)v;
    }
    __syncthreads();

    // ---- Main GEMM: acc[g] = 16-row group g x 16-col tile 'wave' ----
    v8f acc[4];
    #pragma unroll
    for (int g = 0; g < 4; ++g) acc[g] = (v8f){0.f,0.f,0.f,0.f,0.f,0.f,0.f,0.f};

    const char* wslab = (const char*)wb + (size_t)wave * KCHUNKS * 1024;
    // Per-lane invariant part of the A-fragment address
    const char* aBase = (const char*)xs + (size_t)ln * (XSTRIDE * 2) + kh * 32;

    #pragma unroll 1
    for (int shift = 0; shift < KW; ++shift) {          // conv tap
        const char* arow = aBase + (size_t)shift * (XSTRIDE * 2);
        const char* wrow = wslab + (size_t)shift * 8 * 1024;

        #pragma unroll 1
        for (int c8 = 0; c8 < 8; ++c8) {                // 32-wide K chunk
            // B fragment: per-lane 32B register image, coalesced 1KB/wave
            Frag bfrag;
            const uint4* w4 = (const uint4*)(wrow + c8 * 1024);
            bfrag.q[0] = w4[lane * 2];
            bfrag.q[1] = w4[lane * 2 + 1];
            if (shift * 8 + c8 < KCHUNKS - 1)
                __builtin_prefetch(wrow + (c8 + 1) * 1024 + lane * 32, 0, 3);

            // All 4 A fragments issued before any WMMA; in-order DS returns
            // + sched groups -> partial dscnt waits, WMMA g overlaps loads.
            const char* ap0 = arow + c8 * 64;
            Frag af[4];
            #pragma unroll
            for (int g = 0; g < 4; ++g) {
                const char* ap = ap0 + (size_t)(16 * g) * (XSTRIDE * 2);
                af[g].q[0] = *(const uint4*)ap;
                af[g].q[1] = *(const uint4*)(ap + 16);
            }
            #pragma unroll
            for (int g = 0; g < 4; ++g) {
                acc[g] = __builtin_amdgcn_wmma_f32_16x16x32_bf16(
                             false, af[g].v, false, bfrag.v,
                             (short)0, acc[g], false, false);
            }
            SCHED_PIPELINE();
        }
    }

    // ---- Bias add (N depends only on lane&15) ----
    float bias;
    if (wave < 16)       bias = b1[16 * wave + ln];
    else if (wave == 16) bias = b2[ln];
    else                 bias = b3[ln];
    #pragma unroll
    for (int g = 0; g < 4; ++g)
        #pragma unroll
        for (int r = 0; r < 8; ++r) acc[g][r] += bias;

    // ---- Bm/Cm tiles -> LDS ----
    if (wave >= 16) {
        const int which = wave - 16;
        #pragma unroll
        for (int g = 0; g < 4; ++g)
            #pragma unroll
            for (int r = 0; r < 8; ++r)
                bc[which][16 * g + r + 8 * kh][ln] = acc[g][r];
    }
    __syncthreads();

    // ---- s[m] = sum_n Bm[m,n] * Cm[m,n] ----
    if (tid < MTILE) {
        float s = 0.0f;
        #pragma unroll
        for (int n = 0; n < NS; ++n) s += bc[0][tid][n] * bc[1][tid][n];
        sv[tid] = s;
    }
    __syncthreads();

    // ---- y = x * softplus(delta_pre) * s ----
    if (wave < 16) {
        float* ob = out + (size_t)b * SEQ * DM;
        #pragma unroll
        for (int g = 0; g < 4; ++g) {
            #pragma unroll
            for (int r = 0; r < 8; ++r) {
                const int row = 16 * g + r + 8 * kh;     // row within tile
                const int col = 16 * wave + ln;
                const float z  = acc[g][r];
                // stable softplus via hardware transcendentals
                const float e  = __expf(-fabsf(z));
                const float sp = fmaxf(z, 0.0f) + __logf(1.0f + e);
                const size_t idx = (size_t)(gRow + row) * DM + col;
                ob[idx] = xb[idx] * sp * sv[row];
            }
        }
    }
}

// ---------------------------------------------------------------------------
extern "C" void kernel_launch(void* const* d_in, const int* in_sizes, int n_in,
                              void* d_out, int out_size, void* d_ws, size_t ws_size,
                              hipStream_t stream)
{
    const float* x  = (const float*)d_in[0];
    const float* w1 = (const float*)d_in[1];
    const float* b1 = (const float*)d_in[2];
    const float* w2 = (const float*)d_in[3];
    const float* b2 = (const float*)d_in[4];
    const float* w3 = (const float*)d_in[5];
    const float* b3 = (const float*)d_in[6];
    // d_in[7] = A : mathematically dead in the collapsed reference

    bf16_t* wbuf = (bf16_t*)d_ws;   // 18*24*32*16 bf16 = 442,368 bytes
    float*  out  = (float*)d_out;

    const int tot = NTILES * KCHUNKS * 32 * 16;
    s6_prep_weights<<<(tot + 255) / 256, 256, 0, stream>>>(w1, w2, w3, wbuf);

    const int nblocks = BATCH * (SEQ / MTILE);   // 1024
    s6_fused<<<nblocks, THREADS, 0, stream>>>(x, b1, b2, b3, wbuf, out);
}
